// ModelE_44083544326532
// MI455X (gfx1250) — matrix operand order repaired and checked by reading, not compiled, not run on hardware
//
#include <hip/hip_runtime.h>
#include <hip/hip_bf16.h>

typedef __attribute__((ext_vector_type(16))) _Float16 v16h;
typedef __attribute__((ext_vector_type(8)))  _Float16 v8h;
typedef __attribute__((ext_vector_type(8)))  float    v8f;

#define LRELU_SLOPE 0.2f
#define BN_EPS 1e-5f
#define NPTS 1024
#define BATCH 4
#define KNN 16

__device__ __forceinline__ v16h concat8(v8h lo, v8h hi) {
  return __builtin_shufflevector(lo, hi, 0, 1, 2, 3, 4, 5, 6, 7,
                                 8, 9, 10, 11, 12, 13, 14, 15);
}

// A fragment (16-bit A 16x32 layout): two contiguous 16B LDS loads per lane.
__device__ __forceinline__ v16h load_afrag(const _Float16* __restrict__ row,
                                           int kb, int halfw) {
  const v8h* p = (const v8h*)(row + kb);
  return concat8(p[halfw], p[2 + halfw]);
}

// B fragment from column-major Bs: k = halfw*8..+7 and 16+halfw*8..+7 -> two 16B LDS loads.
__device__ __forceinline__ v16h load_bfrag(const _Float16* __restrict__ col,
                                           int halfw) {
  const v8h* p = (const v8h*)col;
  return concat8(p[halfw], p[2 + halfw]);
}

// Async 16B copy global -> LDS (no VGPR round-trip). GVS addressing:
// mem = SGPR64(base) + VGPR32(byte offset). Tracked by ASYNCcnt.
__device__ __forceinline__ void async_copy16(const _Float16* base, int goff_bytes,
                                             const _Float16* lds) {
  asm volatile("global_load_async_to_lds_b128 %0, %1, %2"
               :
               : "v"((unsigned)(unsigned long long)lds),
                 "v"(goff_bytes),
                 "s"(base)
               : "memory");
}

__device__ __forceinline__ void wait_async0() {
  asm volatile("s_wait_asynccnt 0x0" ::: "memory");
}

// ---------------------------------------------------------------------------
// One-time per call: W (K x C, f32 row-major) -> Wt (C x KPAD, f16 col-major,
// zero-padded K). Makes GEMM staging a pure contiguous async block copy.
// ---------------------------------------------------------------------------
__global__ void convert_w_kernel(const float* __restrict__ src,
                                 _Float16* __restrict__ dst,
                                 int K, int C, int KPAD) {
  const int i = blockIdx.x * blockDim.x + threadIdx.x;
  if (i >= C * KPAD) return;
  const int c = i / KPAD;
  const int r = i - c * KPAD;
  dst[i] = (_Float16)((r < K) ? src[r * C + c] : 0.f);
}

// ---------------------------------------------------------------------------
// KNN: one thread per point, 16-deep insertion sort over all 1024 candidates.
// ---------------------------------------------------------------------------
__global__ void knn_kernel(const float* __restrict__ xyz, int* __restrict__ knn_idx) {
  const int b = blockIdx.x >> 2;
  const int n = ((blockIdx.x & 3) << 8) + threadIdx.x;
  const float* pb = xyz + b * NPTS * 3;
  const float px = pb[n * 3 + 0], py = pb[n * 3 + 1], pz = pb[n * 3 + 2];
  float bd[KNN];
  int   bi[KNN];
#pragma unroll
  for (int j = 0; j < KNN; ++j) { bd[j] = 3.4e38f; bi[j] = 0; }
  __shared__ float sx[256], sy[256], sz[256];
  for (int t = 0; t < NPTS; t += 256) {
    __syncthreads();
    sx[threadIdx.x] = pb[(t + threadIdx.x) * 3 + 0];
    sy[threadIdx.x] = pb[(t + threadIdx.x) * 3 + 1];
    sz[threadIdx.x] = pb[(t + threadIdx.x) * 3 + 2];
    __syncthreads();
    for (int m = 0; m < 256; ++m) {
      const float dx = sx[m] - px, dy = sy[m] - py, dz = sz[m] - pz;
      float d = dx * dx + dy * dy + dz * dz;
      int   mi = t + m;
      if (d < bd[KNN - 1]) {
#pragma unroll
        for (int j = 0; j < KNN; ++j) {
          if (d < bd[j]) {
            const float td = bd[j]; const int ti = bi[j];
            bd[j] = d; bi[j] = mi; d = td; mi = ti;
          }
        }
      }
    }
  }
#pragma unroll
  for (int j = 0; j < KNN; ++j) knn_idx[(b * NPTS + n) * KNN + j] = bi[j];
}

// ---------------------------------------------------------------------------
// EdgeConv GEMM + fused max-over-k + bias + BN-stat accumulation.
// PPB points per block, one wave per point. W tiles staged into LDS with
// async b128 copies from the pre-transposed f16 Wt.
// ---------------------------------------------------------------------------
template <int CIN, int COUT, int KTOT, int PPB>
__global__ void edgeconv_wmma(const float*    __restrict__ feat,
                              const int*      __restrict__ knn_idx,
                              const _Float16* __restrict__ Wt,   // [COUT][KTOT]
                              const float*    __restrict__ bias,
                              float* __restrict__ hpre,
                              float* __restrict__ stat_s,
                              float* __restrict__ stat_q) {
  constexpr int AST = KTOT + 8;          // padded LDS stride (halves)
  constexpr int NTHREADS = PPB * 32;
  constexpr int CPC = KTOT / 8;          // 16B copies per column
  __shared__ __align__(16) _Float16 As[PPB * 16 * AST];
  __shared__ __align__(16) _Float16 Bs[16 * AST];

  const int bn0 = blockIdx.x * PPB;
  const int b   = bn0 >> 10;
  const int tid = threadIdx.x;

  // Build edge rows: [center(CIN) | neigh_k - center(CIN) | 0-pad]
  for (int i = tid; i < PPB * 16 * KTOT; i += NTHREADS) {
    const int p   = i / (16 * KTOT);
    const int rem = i - p * (16 * KTOT);
    const int k   = rem / KTOT;
    const int c   = rem - k * KTOT;
    const int bn  = bn0 + p;
    float val = 0.f;
    if (c < CIN) {
      val = feat[bn * CIN + c];
    } else if (c < 2 * CIN) {
      const int cc = c - CIN;
      const int ni = knn_idx[bn * KNN + k];
      val = feat[(b * NPTS + ni) * CIN + cc] - feat[bn * CIN + cc];
    }
    As[(p * 16 + k) * AST + c] = (_Float16)val;
  }

  const int wave  = tid >> 5;
  const int lane  = tid & 31;
  const int halfw = lane >> 4;
  const int mrow  = lane & 15;
  const _Float16* Arow = As + (wave * 16 + mrow) * AST;

  for (int ct = 0; ct < COUT / 16; ++ct) {
    __syncthreads();
    // Async-stage 16 columns of Wt into padded col-major LDS.
    for (int i = tid; i < 16 * CPC; i += NTHREADS) {
      const int c = i / CPC;
      const int j = i - c * CPC;
      async_copy16(Wt, (int)(((ct * 16 + c) * KTOT + j * 8) * sizeof(_Float16)),
                   Bs + c * AST + j * 8);
    }
    wait_async0();
    __syncthreads();

    v8f acc;
#pragma unroll
    for (int r = 0; r < 8; ++r) acc[r] = 0.f;

    for (int kb = 0; kb < KTOT; kb += 32) {
      const v16h afrag = load_afrag(Arow, kb, halfw);
      const v16h bfrag = load_bfrag(Bs + mrow * AST + kb, halfw);
      acc = __builtin_amdgcn_wmma_f32_16x16x32_f16(
          false, afrag, false, bfrag, (short)0, acc, false, false);
    }

    // max over the 16 k-rows (VGPR r holds rows r / r+8 across lane halves)
    float m = acc[0];
#pragma unroll
    for (int r = 1; r < 8; ++r) m = fmaxf(m, acc[r]);
    m = fmaxf(m, __shfl_xor(m, 16, 32));
    if (halfw == 0) {
      const int co  = ct * 16 + mrow;
      const int bn  = bn0 + wave;
      const float v = m + bias[co];
      hpre[bn * COUT + co] = v;
      atomicAdd(&stat_s[co], v);
      atomicAdd(&stat_q[co], v * v);
    }
  }
}

// ---------------------------------------------------------------------------
// Elementwise BatchNorm + LeakyReLU (in place).
// ---------------------------------------------------------------------------
__global__ void bn_lrelu_kernel(float* __restrict__ buf,
                                const float* __restrict__ stat_s,
                                const float* __restrict__ stat_q,
                                const float* __restrict__ gamma,
                                const float* __restrict__ beta,
                                int cout, int total) {
  const int i = blockIdx.x * blockDim.x + threadIdx.x;
  if (i >= total) return;
  const int c = i % cout;
  const float inv  = 1.0f / (float)(BATCH * NPTS);
  const float mean = stat_s[c] * inv;
  const float var  = stat_q[c] * inv - mean * mean;
  float v = (buf[i] - mean) * rsqrtf(var + BN_EPS) * gamma[c] + beta[c];
  buf[i] = (v > 0.f) ? v : LRELU_SLOPE * v;
}

// ---------------------------------------------------------------------------
// Embedding GEMM: (4096 x 256) x (256 x 512) + bias + BN-stat atomics.
// One block per 16-row M tile; 8 waves x 4 column tiles; async W staging.
// ---------------------------------------------------------------------------
__global__ void embed_wmma(const float*    __restrict__ feat,
                           const _Float16* __restrict__ Wet,  // [512][256] f16
                           const float*    __restrict__ bse,
                           float* __restrict__ epre,
                           float* __restrict__ stat_s,
                           float* __restrict__ stat_q) {
  constexpr int KTOT = 256, COUT = 512, TPW = 4, NTHREADS = 256;
  constexpr int AST = KTOT + 8;  // A row stride (halves)
  constexpr int BST = 40;        // B col stride (halves) per 32-K chunk
  __shared__ __align__(16) _Float16 As[16 * AST];     // ~8.3 KB
  __shared__ __align__(16) _Float16 Bs[COUT * BST];   // 40 KB

  const int row0 = blockIdx.x * 16;
  const int tid  = threadIdx.x;

  for (int i = tid; i < 16 * KTOT; i += NTHREADS) {
    const int r = i / KTOT;
    const int c = i - r * KTOT;
    As[r * AST + c] = (_Float16)feat[(row0 + r) * KTOT + c];
  }

  const int wave  = tid >> 5;
  const int lane  = tid & 31;
  const int halfw = lane >> 4;
  const int mrow  = lane & 15;
  const _Float16* Arow = As + mrow * AST;

  v8f acc[TPW];
#pragma unroll
  for (int t = 0; t < TPW; ++t)
#pragma unroll
    for (int r = 0; r < 8; ++r) acc[t][r] = 0.f;

  for (int kb = 0; kb < KTOT; kb += 32) {
    __syncthreads();
    // Async-stage this K chunk of all 512 columns: 4 x 16B per column.
    for (int i = tid; i < COUT * 4; i += NTHREADS) {
      const int c    = i >> 2;
      const int part = i & 3;
      async_copy16(Wet, (int)((c * KTOT + kb + part * 8) * sizeof(_Float16)),
                   Bs + c * BST + part * 8);
    }
    wait_async0();
    __syncthreads();

    const v16h afrag = load_afrag(Arow, kb, halfw);
#pragma unroll
    for (int t = 0; t < TPW; ++t) {
      const int c0 = (wave * TPW + t) * 16;
      const v16h bfrag = load_bfrag(Bs + (c0 + mrow) * BST, halfw);
      acc[t] = __builtin_amdgcn_wmma_f32_16x16x32_f16(
          false, afrag, false, bfrag, (short)0, acc[t], false, false);
    }
  }

#pragma unroll
  for (int t = 0; t < TPW; ++t) {
    const int co = (wave * TPW + t) * 16 + mrow;
    const float bs = bse[co];
    float sum = 0.f, sq = 0.f;
#pragma unroll
    for (int r = 0; r < 8; ++r) {
      const float v = acc[t][r] + bs;
      epre[(row0 + r + 8 * halfw) * COUT + co] = v;
      sum += v;
      sq  += v * v;
    }
    sum += __shfl_xor(sum, 16, 32);
    sq  += __shfl_xor(sq, 16, 32);
    if (halfw == 0) {
      atomicAdd(&stat_s[co], sum);
      atomicAdd(&stat_q[co], sq);
    }
  }
}

// ---------------------------------------------------------------------------
// Final: BN + LeakyReLU + max over N. One warp per (b, channel).
// ---------------------------------------------------------------------------
__global__ void final_max_kernel(const float* __restrict__ epre,
                                 const float* __restrict__ stat_s,
                                 const float* __restrict__ stat_q,
                                 const float* __restrict__ gamma,
                                 const float* __restrict__ beta,
                                 float* __restrict__ out) {
  const int warpId = (blockIdx.x * blockDim.x + threadIdx.x) >> 5;
  const int lane   = threadIdx.x & 31;
  const int b  = warpId >> 9;
  const int co = warpId & 511;
  const float inv  = 1.0f / (float)(BATCH * NPTS);
  const float mean = stat_s[co] * inv;
  const float var  = stat_q[co] * inv - mean * mean;
  const float sc   = rsqrtf(var + BN_EPS) * gamma[co];
  const float bt   = beta[co];
  float m = -3.4e38f;
  for (int n = lane; n < NPTS; n += 32) {
    float v = (epre[(b * NPTS + n) * 512 + co] - mean) * sc + bt;
    v = (v > 0.f) ? v : LRELU_SLOPE * v;
    m = fmaxf(m, v);
  }
#pragma unroll
  for (int off = 16; off; off >>= 1) m = fmaxf(m, __shfl_xor(m, off, 32));
  if (lane == 0) out[b * 512 + co] = m;
}

__global__ void zero_kernel(float* __restrict__ p, int n) {
  const int i = blockIdx.x * blockDim.x + threadIdx.x;
  if (i < n) p[i] = 0.f;
}

// ---------------------------------------------------------------------------
extern "C" void kernel_launch(void* const* d_in, const int* in_sizes, int n_in,
                              void* d_out, int out_size, void* d_ws, size_t ws_size,
                              hipStream_t stream) {
  (void)in_sizes; (void)n_in; (void)out_size; (void)ws_size;

  const float* xyz = (const float*)d_in[0];
  const float* W1  = (const float*)d_in[1];
  const float* b1  = (const float*)d_in[2];
  const float* g1  = (const float*)d_in[3];
  const float* be1 = (const float*)d_in[4];
  const float* W2  = (const float*)d_in[5];
  const float* b2  = (const float*)d_in[6];
  const float* g2  = (const float*)d_in[7];
  const float* be2 = (const float*)d_in[8];
  const float* W3  = (const float*)d_in[9];
  const float* b3  = (const float*)d_in[10];
  const float* g3  = (const float*)d_in[11];
  const float* be3 = (const float*)d_in[12];
  const float* We  = (const float*)d_in[13];
  const float* bse = (const float*)d_in[14];
  const float* ge  = (const float*)d_in[15];
  const float* bee = (const float*)d_in[16];
  float* out = (float*)d_out;

  // workspace layout
  char*  ws    = (char*)d_ws;
  int*   idx   = (int*)ws;                                   // 65536 ints
  float* buf1  = (float*)(ws + 65536 * sizeof(int));         // 4096 x 64
  float* buf2  = buf1 + 4096 * 64;                           // 4096 x 128
  float* buf3  = buf2 + 4096 * 128;                          // 4096 x 256
  float* bufe  = buf3 + 4096 * 256;                          // 4096 x 512
  float* stats = bufe + 4096 * 512;                          // 1920 floats
  float* s1 = stats;        float* q1 = s1 + 64;
  float* s2 = q1 + 64;      float* q2 = s2 + 128;
  float* s3 = q2 + 128;     float* q3 = s3 + 256;
  float* se = q3 + 256;     float* qe = se + 512;
  _Float16* w1t = (_Float16*)(stats + 1920);   // [64][32]
  _Float16* w2t = w1t + 64 * 32;               // [128][128]
  _Float16* w3t = w2t + 128 * 128;             // [256][256]
  _Float16* wet = w3t + 256 * 256;             // [512][256]

  zero_kernel<<<(1920 + 255) / 256, 256, 0, stream>>>(stats, 1920);
  knn_kernel<<<BATCH * 4, 256, 0, stream>>>(xyz, idx);

  convert_w_kernel<<<(64 * 32 + 255) / 256, 256, 0, stream>>>(W1, w1t, 6, 64, 32);
  convert_w_kernel<<<(128 * 128 + 255) / 256, 256, 0, stream>>>(W2, w2t, 128, 128, 128);
  convert_w_kernel<<<(256 * 256 + 255) / 256, 256, 0, stream>>>(W3, w3t, 256, 256, 256);
  convert_w_kernel<<<(512 * 256 + 255) / 256, 256, 0, stream>>>(We, wet, 256, 512, 256);

  edgeconv_wmma<3, 64, 32, 8><<<BATCH * NPTS / 8, 256, 0, stream>>>(
      xyz, idx, w1t, b1, buf1, s1, q1);
  bn_lrelu_kernel<<<(4096 * 64 + 255) / 256, 256, 0, stream>>>(
      buf1, s1, q1, g1, be1, 64, 4096 * 64);

  edgeconv_wmma<64, 128, 128, 8><<<BATCH * NPTS / 8, 256, 0, stream>>>(
      buf1, idx, w2t, b2, buf2, s2, q2);
  bn_lrelu_kernel<<<(4096 * 128 + 255) / 256, 256, 0, stream>>>(
      buf2, s2, q2, g2, be2, 128, 4096 * 128);

  edgeconv_wmma<128, 256, 256, 4><<<BATCH * NPTS / 4, 128, 0, stream>>>(
      buf2, idx, w3t, b3, buf3, s3, q3);
  bn_lrelu_kernel<<<(4096 * 256 + 255) / 256, 256, 0, stream>>>(
      buf3, s3, q3, g3, be3, 256, 4096 * 256);

  embed_wmma<<<256, 256, 0, stream>>>(buf3, wet, bse, bufe, se, qe);

  final_max_kernel<<<(BATCH * 512 * 32) / 256, 256, 0, stream>>>(
      bufe, se, qe, ge, bee, out);
}
